// CTRNN_1219770712302
// MI455X (gfx1250) — matrix operand order repaired and checked by reading, not compiled, not run on hardware
//
#include <hip/hip_runtime.h>

// ---------------- problem constants (match reference) ----------------
static constexpr int T = 256;
static constexpr int B = 64;
static constexpr int I = 512;
static constexpr int H = 2048;
static constexpr float ALPHA = 0.01f;

// ---------------- CDNA5 WMMA / TDM types ----------------
typedef __attribute__((ext_vector_type(16))) __bf16       v16bf;
typedef __attribute__((ext_vector_type(8)))  float        v8f;
typedef __attribute__((ext_vector_type(8)))  unsigned int v8u;
typedef __attribute__((ext_vector_type(4)))  unsigned int u32x4;
typedef __attribute__((ext_vector_type(8)))  int          i32x8;
typedef __attribute__((ext_vector_type(4)))  int          i32x4;

// native-cast pair pack: lets clang use packed bf16 cvt ops
__device__ __forceinline__ unsigned int pack2_bf16(float a, float b) {
    unsigned short lo = __builtin_bit_cast(unsigned short, (__bf16)a);
    unsigned short hi = __builtin_bit_cast(unsigned short, (__bf16)b);
    return (unsigned int)lo | ((unsigned int)hi << 16);
}

// =====================================================================
// Pack kernel: W (row-major [N=H][K], fp32) -> bf16 B-fragments in the
// per-lane B layout this file feeds v_wmma_f32_16x16x32_bf16 (contiguous k, same as A).
// frag layout: [nTile][kTile][lane(32)][16 bf16]
//   lane L: n = nTile*16 + (L&15); elem j: k = kTile*32 + (L>>4)*16 + j
// =====================================================================
__global__ __launch_bounds__(256)
void pack_b_frags(const float* __restrict__ W,
                  unsigned short* __restrict__ frag,
                  int K)
{
    const int kTiles = K >> 5;
    const int totalLanes = (H / 16) * kTiles * 32;
    int tid = blockIdx.x * blockDim.x + threadIdx.x;
    if (tid >= totalLanes) return;

    const int lane  = tid & 31;
    const int tile  = tid >> 5;
    const int kTile = tile % kTiles;
    const int nTile = tile / kTiles;
    const int n     = nTile * 16 + (lane & 15);
    const int kBase = kTile * 32 + (lane >> 4) * 16;

    const float* src = W + (size_t)n * K + kBase;
    unsigned int packed[8];
#pragma unroll
    for (int p = 0; p < 8; ++p)
        packed[p] = pack2_bf16(src[2 * p + 0], src[2 * p + 1]);

    uint4* dst = (uint4*)(frag + (size_t)tid * 16);
    dst[0] = make_uint4(packed[0], packed[1], packed[2], packed[3]);
    dst[1] = make_uint4(packed[4], packed[5], packed[6], packed[7]);
}

// =====================================================================
// TDM: load a 2-D fp32 tile [rows=64][cols=32] (row stride ldA elements)
// from global into LDS at byte offset lds_off, with pad_amount=4 dwords
// after every 32 dwords -> LDS row stride 36 dwords (bank-conflict-free).
// D# per CDNA5 ISA (group0 + group1; 2-D => groups
// 2/3 unused, zero-filled; 6-arg builtin form of this toolchain).
// =====================================================================
__device__ __forceinline__ void tdm_load_panel(unsigned lds_off,
                                               const float* gptr,
                                               unsigned ldA)
{
    unsigned long long ga = (unsigned long long)(uintptr_t)gptr;

    u32x4 g0;
    g0[0] = 1u;                                   // count=1 (valid), user mode
    g0[1] = lds_off;                              // lds_addr (bytes)
    g0[2] = (unsigned)(ga & 0xFFFFFFFFu);         // global_addr[31:0]
    g0[3] = (unsigned)((ga >> 32) & 0x1FFFFFFu)   // global_addr[56:32]
          | (2u << 30);                           // type = 2 ("image")

    i32x8 g1;
    g1[0] = (int)((2u << 16)        // data_size = 4 B
                | (1u << 20)        // pad_enable
                | (4u << 22)        // pad_interval: 2<<4 = 32 dwords
                | (3u << 25));      // pad_amount: 3+1 = 4 dwords
    g1[1] = (int)((ldA & 0xFFFFu) << 16);                   // tensor_dim0 lo
    g1[2] = (int)(((ldA >> 16) & 0xFFFFu) | (64u << 16));   // dim0 hi | tensor_dim1=64
    g1[3] = (int)(32u << 16);                               // tile_dim0 = 32
    g1[4] = (int)64;                                        // tile_dim1 = 64
    g1[5] = (int)ldA;                                       // tensor_dim0_stride lo
    g1[6] = 0;                                              // stride hi / dim1_stride
    g1[7] = 0;

    i32x4 z4 = {0, 0, 0, 0};
    i32x8 z8 = {0, 0, 0, 0, 0, 0, 0, 0};
    __builtin_amdgcn_tensor_load_to_lds(g0, g1, z4, z4, z8, 0);
}

// =====================================================================
// One recurrence step:
//   acc[64,2048] = x_t @ W_ih^T + h_prev @ W_hh^T + b_ih + b_hh
//   out_t        = (1-ALPHA)*h_prev + ALPHA*relu(acc)
// Block: 128 threads (4 waves); block owns N=32 (2 tiles), all M=64.
// Wave w -> M-tile w; 2 C tiles per wave share one A fragment.
// A panel staged fp32 via TDM, double-buffered; bf16 convert at frag read.
// =====================================================================
__global__ __launch_bounds__(128)
void ctrnn_step(const float* __restrict__ x_t,               // [B][I]
                const float* __restrict__ h_prev,            // [B][H] or null
                const unsigned short* __restrict__ whh_frag,
                const unsigned short* __restrict__ wih_frag,
                const float* __restrict__ b_ih,
                const float* __restrict__ b_hh,
                float* __restrict__ out_t)                   // [B][H]
{
    constexpr int ROW_DW = 36;                   // 32 data + 4 pad dwords
    __shared__ float sA[2][64 * ROW_DW];         // 2 x 9216 B

    const int tid   = threadIdx.x;
    const int lane  = tid & 31;
    const int wave  = tid >> 5;                  // M-tile
    const int g     = lane >> 4;
    const int nCol  = lane & 15;
    const int nBase = blockIdx.x * 32;

    v8f acc0, acc1;
    {
        float bias0 = b_ih[nBase + nCol]      + b_hh[nBase + nCol];
        float bias1 = b_ih[nBase + 16 + nCol] + b_hh[nBase + 16 + nCol];
#pragma unroll
        for (int r = 0; r < 8; ++r) { acc0[r] = bias0; acc1[r] = bias1; }
    }

    const unsigned ldsOff0 = (unsigned)(uintptr_t)&sA[0][0]; // low 32b = LDS addr
    const unsigned ldsOff1 = (unsigned)(uintptr_t)&sA[1][0];

    auto run_panel = [&](const float* __restrict__ A, unsigned ldA, int K,
                         const unsigned short* __restrict__ bfrag) {
        const int kTiles = K >> 5;
        if (wave == 0) tdm_load_panel(ldsOff0, A, ldA);      // prologue DMA

        for (int kt = 0; kt < kTiles; ++kt) {
            if (wave == 0) {
                if (kt + 1 < kTiles) {
                    tdm_load_panel((kt & 1) ? ldsOff0 : ldsOff1,
                                   A + (size_t)(kt + 1) * 32, ldA);
                    __builtin_amdgcn_s_wait_tensorcnt(1);    // chunk kt landed
                } else {
                    __builtin_amdgcn_s_wait_tensorcnt(0);
                }
            }
            __syncthreads();                                 // panel visible

            const float* panel = &sA[kt & 1][0];
            // ---- A fragment: lane = row m; elems 0-7 -> K=g*16+e,
            //                  elems 8-15 -> K=g*16+8+e (contiguous, matches B pack)
            const int m = wave * 16 + nCol;
            const float4* p0 = (const float4*)(panel + m * ROW_DW + g * 16);
            const float4* p1 = (const float4*)(panel + m * ROW_DW + g * 16 + 8);
            float4 fa = p0[0], fb = p0[1];
            float4 fc = p1[0], fd = p1[1];
            v8u au;
            au[0] = pack2_bf16(fa.x, fa.y); au[1] = pack2_bf16(fa.z, fa.w);
            au[2] = pack2_bf16(fb.x, fb.y); au[3] = pack2_bf16(fb.z, fb.w);
            au[4] = pack2_bf16(fc.x, fc.y); au[5] = pack2_bf16(fc.z, fc.w);
            au[6] = pack2_bf16(fd.x, fd.y); au[7] = pack2_bf16(fd.z, fd.w);
            v16bf Af = __builtin_bit_cast(v16bf, au);

            // ---- two B fragments (packed, L2-resident), two WMMAs
#pragma unroll
            for (int j = 0; j < 2; ++j) {
                size_t nTile = (size_t)(blockIdx.x * 2 + j);
                const uint4* bp = (const uint4*)(bfrag +
                    ((nTile * kTiles + kt) * 32 + lane) * 16);
                uint4 b0 = bp[0];
                uint4 b1 = bp[1];
                if (kt + 1 < kTiles)
                    __builtin_prefetch((const void*)(bfrag +
                        ((nTile * kTiles + (kt + 1)) * 32 + lane) * 16), 0, 1);
                v8u bu;
                bu[0] = b0.x; bu[1] = b0.y; bu[2] = b0.z; bu[3] = b0.w;
                bu[4] = b1.x; bu[5] = b1.y; bu[6] = b1.z; bu[7] = b1.w;
                v16bf Bf = __builtin_bit_cast(v16bf, bu);
                if (j == 0)
                    acc0 = __builtin_amdgcn_wmma_f32_16x16x32_bf16(
                        false, Af, false, Bf, (short)0, acc0, false, false);
                else
                    acc1 = __builtin_amdgcn_wmma_f32_16x16x32_bf16(
                        false, Af, false, Bf, (short)0, acc1, false, false);
            }
            __syncthreads();     // all reads done before buffer re-DMA
        }
    };

    if (h_prev) run_panel(h_prev, H, H, whh_frag);   // recurrent GEMM (t>0)
    run_panel(x_t, I, I, wih_frag);                  // input projection

    // -------- epilogue: leaky mix + relu, write h_t --------
#pragma unroll
    for (int r = 0; r < 8; ++r) {
        int m = wave * 16 + g * 8 + r;
#pragma unroll
        for (int j = 0; j < 2; ++j) {
            int n = nBase + j * 16 + nCol;
            size_t idx = (size_t)m * H + n;
            float hp = h_prev ? h_prev[idx] : 0.0f;
            float a  = (j == 0) ? acc0[r] : acc1[r];
            float hn = a > 0.0f ? a : 0.0f;
            out_t[idx] = (1.0f - ALPHA) * hp + ALPHA * hn;
        }
    }
}

// =====================================================================
extern "C" void kernel_launch(void* const* d_in, const int* in_sizes, int n_in,
                              void* d_out, int out_size, void* d_ws, size_t ws_size,
                              hipStream_t stream)
{
    const float* x    = (const float*)d_in[0];  // [T][B][I]
    const float* W_ih = (const float*)d_in[1];  // [H][I]
    const float* b_ih = (const float*)d_in[2];  // [H]
    const float* W_hh = (const float*)d_in[3];  // [H][H]
    const float* b_hh = (const float*)d_in[4];  // [H]
    float* out = (float*)d_out;                 // [T][B][H] ++ [B][H]

    unsigned short* whh_frag = (unsigned short*)d_ws;            // 8 MB
    unsigned short* wih_frag = whh_frag + (size_t)H * H;         // 2 MB

    {
        int lanesWhh = (H / 16) * (H / 32) * 32;
        pack_b_frags<<<(lanesWhh + 255) / 256, 256, 0, stream>>>(W_hh, whh_frag, H);
        int lanesWih = (H / 16) * (I / 32) * 32;
        pack_b_frags<<<(lanesWih + 255) / 256, 256, 0, stream>>>(W_ih, wih_frag, I);
    }

    // sequential scan over time; stream order provides the h dependency
    for (int t = 0; t < T; ++t) {
        const float* h_prev = (t == 0) ? nullptr : out + (size_t)(t - 1) * B * H;
        ctrnn_step<<<H / 32, 128, 0, stream>>>(
            x + (size_t)t * B * I, h_prev, whh_frag, wih_frag,
            b_ih, b_hh, out + (size_t)t * B * H);
    }

    // h_last = out[T-1]
    (void)hipMemcpyAsync(out + (size_t)T * B * H, out + (size_t)(T - 1) * B * H,
                         (size_t)B * H * sizeof(float), hipMemcpyDeviceToDevice,
                         stream);
}
